// ConvDeepSet_54683523612790
// MI455X (gfx1250) — matrix operand order repaired and checked by reading, not compiled, not run on hardware
//
#include <hip/hip_runtime.h>

typedef __attribute__((ext_vector_type(16))) _Float16 v16h;
typedef __attribute__((ext_vector_type(8)))  _Float16 v8h;
typedef __attribute__((ext_vector_type(8)))  float    v8f;

union Frag16 { v16h v; v8h h[2]; };

#define N_CTX   2048
#define BATCH   16
#define M_TGT   4096
#define COUT    64
#define RDIM    128
#define MTILE   128
#define NCHUNK  64
#define ROW_STRIDE 72   // halves per row (144B): 16B-aligned fragment offsets, 16 distinct banks
#define AT_STRIDE  72
#define WT_STRIDE  72

__launch_bounds__(256)
__global__ void convdeepset_fused_kernel(const float* __restrict__ ctx_in,
                                         const float* __restrict__ ctx_out,
                                         const float* __restrict__ tgt_in,
                                         const float* __restrict__ lsp,
                                         const float* __restrict__ W,
                                         const float* __restrict__ bias,
                                         float* __restrict__ out)
{
    __shared__ float xs[N_CTX];                 // context x for this batch
    __shared__ float ys[MTILE];                 // target x tile
    __shared__ float densP[256];                // per-thread density partials
    __shared__ float densL[MTILE];              // reduced density per m
    __shared__ __align__(16) _Float16 Kt[2][MTILE * ROW_STRIDE];  // K tile [m][n 0..63] f16
    __shared__ __align__(16) _Float16 coT[2][COUT * ROW_STRIDE];  // co chunk [c][n 0..63] f16
    __shared__ __align__(16) _Float16 At[MTILE * AT_STRIDE];      // normalized tile [m][c]
    __shared__ __align__(16) _Float16 WtT[RDIM * WT_STRIDE];      // W[:,1:] as [r][c] f16
    __shared__ float w0L[RDIM];
    __shared__ float biasL[RDIM];

    const int tid   = threadIdx.x;
    const int wid   = tid >> 5;                  // wave id 0..7 == m-subtile
    const int lane  = tid & 31;
    const int lrow  = lane & 15;
    const int lhalf = lane >> 4;

    const int b      = blockIdx.x >> 5;          // batch index (M/128 = 32 tiles)
    const int m_base = (blockIdx.x & 31) * MTILE;

    const float ls  = lsp[0];
    const float cl2 = (-0.5f / (ls * ls)) * 1.44269504088896340736f; // fold log2(e)

    // ---------------- Phase 0: stage invariants into LDS ----------------
    for (int i = tid; i < N_CTX; i += 256)
        xs[i] = ctx_in[i * BATCH + b];                      // (N,B,1) layout
    if (tid < MTILE)
        ys[tid] = tgt_in[(m_base + tid) * BATCH + b];       // (M,B,1) layout
    for (int i = tid; i < RDIM * COUT; i += 256) {
        int r = i >> 6, c = i & 63;                         // coalesced along c
        WtT[r * WT_STRIDE + c] = (_Float16)W[r * 65 + 1 + c];
    }
    if (tid < RDIM) {
        w0L[tid]   = W[tid * 65];                           // density-channel weight
        biasL[tid] = bias[tid];
    }
    __syncthreads();

    v8f acc[4] = {};                // one m-subtile x 4 channel tiles per wave

    // K-compute mapping: thread -> (m, 32 consecutive n); co mapping: thread -> (c, 16 n)
    const int kcm = tid >> 1;       // m 0..127
    const int kcn = (tid & 1) * 32; // n0 in {0,32}
    const int con = tid >> 6;       // n-group 0..3 (16 n each)
    const int coc = tid & 63;       // channel

    // hoisted global base for co loads; 16 loads at immediate offsets j*4096B
    const float* gco = ctx_out + ((size_t)(con * 16) * BATCH + b) * COUT + coc;

    float dsum = 0.0f;              // deterministic density partial (register)

    for (int ch = 0; ch < N_CTX / NCHUNK; ++ch) {
        const int buf   = ch & 1;
        const int nbase = ch * NCHUNK;

        // ---- produce K tile: 32 exp2 per thread, four ds_store_b128 ----
        {
            const float y = ys[kcm];
            _Float16* kdst = &Kt[buf][kcm * ROW_STRIDE + kcn];
            #pragma unroll
            for (int g = 0; g < 4; ++g) {
                v8h kp;
                #pragma unroll
                for (int j = 0; j < 8; ++j) {
                    float d = xs[nbase + kcn + g * 8 + j] - y;
                    float k = __builtin_amdgcn_exp2f(cl2 * d * d); // v_exp_f32 (TRANS)
                    dsum += k;
                    kp[j] = (_Float16)k;
                }
                *(v8h*)(kdst + g * 8) = kp;
            }
        }
        // ---- load + transpose co chunk to [c][n] f16 (two b128 LDS stores) ----
        {
            v8h cp0, cp1;
            #pragma unroll
            for (int j = 0; j < 8; ++j) {
                cp0[j] = (_Float16)gco[(size_t)j * (BATCH * COUT)];
                cp1[j] = (_Float16)gco[(size_t)(j + 8) * (BATCH * COUT)];
            }
            _Float16* cdst = &coT[buf][coc * ROW_STRIDE + con * 16];
            *(v8h*)cdst       = cp0;
            *(v8h*)(cdst + 8) = cp1;
            if (ch + 1 < N_CTX / NCHUNK)
                __builtin_prefetch(gco + (size_t)NCHUNK * BATCH * COUT, 0, 1);
            gco += (size_t)NCHUNK * BATCH * COUT;
        }
        __syncthreads();

        // ---- WMMA phase: 2 k-steps x 4 channel tiles = 8 WMMAs per wave ----
        #pragma unroll
        for (int ks = 0; ks < 2; ++ks) {
            Frag16 a;
            const v8h* ap = (const v8h*)&Kt[buf][(wid * 16 + lrow) * ROW_STRIDE
                                                 + ks * 32 + lhalf * 8];
            a.h[0] = ap[0];  a.h[1] = ap[2];      // K = {0..7,16..23} / {8..15,24..31}
            #pragma unroll
            for (int ct = 0; ct < 4; ++ct) {
                Frag16 bb;
                const v8h* bp = (const v8h*)&coT[buf][(ct * 16 + lrow) * ROW_STRIDE
                                                      + ks * 32 + lhalf * 8];
                bb.h[0] = bp[0]; bb.h[1] = bp[2];
                acc[ct] = __builtin_amdgcn_wmma_f32_16x16x32_f16(false, a.v, false, bb.v,
                                                                 (short)0, acc[ct],
                                                                 false, false);
            }
        }
    }

    // ---------------- density reduction (deterministic) ----------------
    densP[tid] = dsum;
    __syncthreads();
    if (tid < MTILE)
        densL[tid] = densP[tid * 2] + densP[tid * 2 + 1];
    __syncthreads();

    // ---------------- normalize and park tile as f16 A-matrix ----------------
    #pragma unroll
    for (int v = 0; v < 8; ++v) {
        int m = wid * 16 + v + lhalf * 8;         // C layout: M = vgpr + 8*(lane>=16)
        float inv = __builtin_amdgcn_rcpf(densL[m] + 1e-8f);
        #pragma unroll
        for (int ct = 0; ct < 4; ++ct)
            At[m * AT_STRIDE + ct * 16 + lrow] = (_Float16)(acc[ct][v] * inv);
    }
    __syncthreads();

    // ------------- projection: (128m x 64c) @ (64c x 128r), two passes -------------
    #pragma unroll
    for (int pass = 0; pass < 2; ++pass) {
        v8f p[4] = {};
        #pragma unroll
        for (int kc = 0; kc < 2; ++kc) {
            Frag16 a;
            const v8h* ap = (const v8h*)&At[(wid * 16 + lrow) * AT_STRIDE
                                            + kc * 32 + lhalf * 8];
            a.h[0] = ap[0]; a.h[1] = ap[2];
            #pragma unroll
            for (int j = 0; j < 4; ++j) {
                int rt = pass * 4 + j;
                Frag16 bw;
                const v8h* bp = (const v8h*)&WtT[(rt * 16 + lrow) * WT_STRIDE
                                                 + kc * 32 + lhalf * 8];
                bw.h[0] = bp[0]; bw.h[1] = bp[2];
                p[j] = __builtin_amdgcn_wmma_f32_16x16x32_f16(false, a.v, false, bw.v,
                                                              (short)0, p[j], false, false);
            }
        }
        // epilogue: + dens*W[:,0] + bias, store (M,B,R) f32
        #pragma unroll
        for (int j = 0; j < 4; ++j) {
            int r = (pass * 4 + j) * 16 + lrow;
            float wr0 = w0L[r], br = biasL[r];
            #pragma unroll
            for (int v = 0; v < 8; ++v) {
                int m = wid * 16 + v + lhalf * 8;
                float val = p[j][v] + densL[m] * wr0 + br;
                out[((size_t)(m_base + m) * BATCH + b) * RDIM + r] = val;
            }
        }
    }
}

extern "C" void kernel_launch(void* const* d_in, const int* in_sizes, int n_in,
                              void* d_out, int out_size, void* d_ws, size_t ws_size,
                              hipStream_t stream) {
    (void)in_sizes; (void)n_in; (void)d_ws; (void)ws_size; (void)out_size;
    const float* ctx_in  = (const float*)d_in[0];   // (N,B,1)
    const float* ctx_out = (const float*)d_in[1];   // (N,B,64)
    const float* tgt_in  = (const float*)d_in[2];   // (M,B,1)
    const float* lsp     = (const float*)d_in[3];   // (1,)
    const float* W       = (const float*)d_in[4];   // (128,65)
    const float* bias    = (const float*)d_in[5];   // (128,)
    float* out = (float*)d_out;                     // (M,B,128) f32

    dim3 grid(BATCH * (M_TGT / MTILE));             // 16 * 32 = 512 workgroups
    convdeepset_fused_kernel<<<grid, 256, 0, stream>>>(ctx_in, ctx_out, tgt_in,
                                                       lsp, W, bias, out);
}
